// ContextViTv57_29025388986797
// MI455X (gfx1250) — matrix operand order repaired and checked by reading, not compiled, not run on hardware
//
#include <hip/hip_runtime.h>

// ---------------- model constants (ViT-Base, context-ViT variant) -----------
static constexpr int BB   = 32;    // batch
static constexpr int DD   = 768;   // model dim
static constexpr int HH   = 12;    // heads
static constexpr int DH   = 64;    // head dim
static constexpr int GRID = 14;
static constexpr int PP   = 196;   // patches
static constexpr int NN   = 197;   // tokens
static constexpr int KC   = 50;    // context tokens
static constexpr int HID  = 3072;
static constexpr int DEPTH= 12;

// staged bf16 weight offsets (elements) inside per-layer staging buffer
static constexpr size_t W_Q   = 0;
static constexpr size_t W_KV  = 589824;            // 768*768
static constexpr size_t W_OUT = W_KV + 1179648;    // + 1536*768
static constexpr size_t W_FC1 = W_OUT + 589824;
static constexpr size_t W_FC2 = W_FC1 + 2359296;   // + 3072*768
static constexpr size_t W_LG  = W_FC2 + 2359296;
static constexpr size_t W_TOT = W_LG + 768;        // 7,078,656 elements

typedef __attribute__((ext_vector_type(8)))  __bf16 bf16x8;
typedef __attribute__((ext_vector_type(16))) __bf16 bf16x16;
typedef __attribute__((ext_vector_type(8)))  float  f32x8;

__device__ __forceinline__ unsigned short f2bf(float f) {
  unsigned int u = __float_as_uint(f);
  u += 0x7FFFu + ((u >> 16) & 1u);          // round-to-nearest-even
  return (unsigned short)(u >> 16);
}
__device__ __forceinline__ float bf2f(unsigned short h) {
  return __uint_as_float(((unsigned int)h) << 16);
}
__device__ __forceinline__ float gelu_exact(float v) {
  return 0.5f * v * (1.0f + erff(v * 0.70710678118654752f));
}

// CDNA5 async copy: 16B/lane global -> LDS, tracked by ASYNCcnt (ISA 08 §4).
__device__ __forceinline__ void async_ld16(unsigned lds_off, unsigned long long gaddr) {
  asm volatile("global_load_async_to_lds_b128 %0, %1, off"
               :: "v"(lds_off), "v"(gaddr) : "memory");
}

// ---------------- fp32 -> bf16 conversion (weight staging) ------------------
__global__ void cvt_f32_bf16(const float* __restrict__ src,
                             unsigned short* __restrict__ dst, int n) {
  int i = blockIdx.x * blockDim.x + threadIdx.x;
  if (i < n) dst[i] = f2bf(src[i]);
}

// ---------------- im2col: x_img -> patches bf16 [B*P, 768] ------------------
__global__ void im2col_patches(const float* __restrict__ img,
                               unsigned short* __restrict__ out, int total) {
  int i = blockIdx.x * blockDim.x + threadIdx.x;
  if (i >= total) return;
  int col = i % DD;          // c*256 + a*16 + w
  int r   = i / DD;          // b*196 + p
  int p   = r % PP;
  int b   = r / PP;
  int ch  = col >> 8;
  int a   = (col >> 4) & 15;
  int w   = col & 15;
  int gi  = p / GRID, gj = p % GRID;
  float v = img[(((size_t)b * 3 + ch) * 224 + gi * 16 + a) * 224 + gj * 16 + w];
  out[i] = f2bf(v);
}

// ---------------- cls row init: x[b,0,:] = cls + pos[0] ---------------------
__global__ void init_cls(const float* __restrict__ cls,
                         const float* __restrict__ pos,
                         float* __restrict__ x) {
  int i = blockIdx.x * blockDim.x + threadIdx.x;   // B*768
  if (i >= BB * DD) return;
  int c = i % DD, b = i / DD;
  x[(size_t)b * NN * DD + c] = cls[c] + pos[c];
}

// ---------------- WMMA bf16 GEMM: C[M,N] = A[M,K] @ W[N,K]^T ----------------
// Block tile 128x64 (8 waves, wave tile 16x64 = 4 accumulators).
// A/B tiles double-buffered in LDS, filled with global_load_async_to_lds_b128
// (ASYNCcnt) so the fetch of step s+1 overlaps the WMMAs of step s.
// EPI: 0 = f32 (+bias)            1 = f32 + bias + residual (in-place ok)
//      2 = f32 + bias + pos, row remap (patch embed)  3 = bf16 gelu(bias+acc)
template <int EPI>
__global__ __launch_bounds__(256) void gemm_bf16_wmma(
    const unsigned short* __restrict__ A, const unsigned short* __restrict__ W,
    const float* __restrict__ bias, const float* __restrict__ res,
    const float* __restrict__ pos, float* __restrict__ outF,
    unsigned short* __restrict__ outH, int M, int N, int K,
    int row_div, int row_off) {
  __shared__ unsigned short shA[2][128][32];   // 2 x 8KB
  __shared__ unsigned short shB[2][64][32];    // 2 x 4KB

  const int tid  = threadIdx.x;
  const int wave = tid >> 5;
  const int lane = tid & 31;
  const int lh   = lane >> 4;     // half-wave select
  const int l16  = lane & 15;
  const int m0b  = blockIdx.x * 128;
  const int m0   = m0b + wave * 16;
  const int n0   = blockIdx.y * 64;

  const unsigned ldsA = (unsigned)(uintptr_t)&shA[0][0][0];
  const unsigned ldsB = (unsigned)(uintptr_t)&shB[0][0][0];

  // staging thread mapping (whole block participates every step)
  const int bcol = tid >> 2, bch = tid & 3;                 // B: 64 cols x 4 chunks
  const unsigned long long gB0 =
      (unsigned long long)(uintptr_t)(W + (size_t)(n0 + bcol) * K + bch * 8);
  int arow0 = (tid * 2) >> 2, ach0 = (tid * 2) & 3;         // A: 128 rows x 4 chunks
  int arow1 = (tid * 2 + 1) >> 2, ach1 = (tid * 2 + 1) & 3;
  int gr0 = m0b + arow0; if (gr0 > M - 1) gr0 = M - 1;      // clamp ragged M edge
  int gr1 = m0b + arow1; if (gr1 > M - 1) gr1 = M - 1;
  const unsigned long long gA0 =
      (unsigned long long)(uintptr_t)(A + (size_t)gr0 * K + ach0 * 8);
  const unsigned long long gA1 =
      (unsigned long long)(uintptr_t)(A + (size_t)gr1 * K + ach1 * 8);
  const unsigned lB = ldsB + (unsigned)(bcol * 64 + bch * 16);
  const unsigned lA0 = ldsA + (unsigned)(arow0 * 64 + ach0 * 16);
  const unsigned lA1 = ldsA + (unsigned)(arow1 * 64 + ach1 * 16);

  f32x8 acc[4];
#pragma unroll
  for (int j = 0; j < 4; ++j)
#pragma unroll
    for (int r = 0; r < 8; ++r) acc[j][r] = 0.0f;

  const int nsteps = K >> 5;
  // prologue: stage step 0 into buffer 0  (3 async ops per wave per stage)
  {
    unsigned long long kb = 0;
    async_ld16(lB, gB0 + kb);
    async_ld16(lA0, gA0 + kb);
    async_ld16(lA1, gA1 + kb);
  }
  for (int s = 0; s < nsteps; ++s) {
    const int cur = s & 1;
    const bool more = (s + 1) < nsteps;
    if (more) {                       // stage next step into the other buffer
      unsigned long long kb = (unsigned long long)(s + 1) * 64;  // 32 bf16 = 64B
      unsigned swz = (unsigned)((cur ^ 1) ? 1 : 0);
      async_ld16(lB + swz * 4096, gB0 + kb);
      async_ld16(lA0 + swz * 8192, gA0 + kb);
      async_ld16(lA1 + swz * 8192, gA1 + kb);
      asm volatile("s_wait_asynccnt 0x3" ::: "memory");  // current tile landed
    } else {
      asm volatile("s_wait_asynccnt 0x0" ::: "memory");
    }
    __syncthreads();

    // A-frag (16x32 bf16): lanes 0-15 K{0-15}, lanes 16-31 K{16-31}
    bf16x16 afrag = *reinterpret_cast<const bf16x16*>(&shA[cur][wave * 16 + l16][16 * lh]);
#pragma unroll
    for (int j = 0; j < 4; ++j) {     // B-frag: column = lane%16, 16 K values
      bf16x16 bfrag = *reinterpret_cast<const bf16x16*>(&shB[cur][j * 16 + l16][16 * lh]);
      acc[j] = __builtin_amdgcn_wmma_f32_16x16x32_bf16(
          false, afrag, false, bfrag, (short)0, acc[j], false, false);
    }
    __syncthreads();                  // tile fully consumed before re-stage
  }

  if (m0 >= M) return;                // ragged edge: skip store only
#pragma unroll
  for (int j = 0; j < 4; ++j) {
#pragma unroll
    for (int r = 0; r < 8; ++r) {
      int row = m0 + r + 8 * lh;      // C/D layout: vgpr r -> M = r + 8*lh
      int col = n0 + j * 16 + l16;
      float v = acc[j][r];
      if (bias) v += bias[col];
      if (EPI == 3) {
        outH[(size_t)row * N + col] = f2bf(gelu_exact(v));
      } else if (EPI == 1) {
        size_t o = (size_t)row * N + col;
        outF[o] = v + res[o];
      } else if (EPI == 2) {
        int qd = row / row_div;
        int rm = row - qd * row_div;
        int orow = row + qd + row_off;                 // skip cls slot per batch
        outF[(size_t)orow * N + col] = v + pos[(size_t)(rm + row_off) * N + col];
      } else {
        outF[(size_t)row * N + col] = v;
      }
    }
  }
}

// ---------------- LayerNorm (D=768) -> bf16 ---------------------------------
__global__ __launch_bounds__(256) void ln_bf16(const float* __restrict__ x,
                                               const float* __restrict__ w,
                                               const float* __restrict__ b,
                                               unsigned short* __restrict__ out) {
  __shared__ float red[256];
  const int row = blockIdx.x, tid = threadIdx.x;
  const float* xr = x + (size_t)row * DD;
  float v0 = xr[tid], v1 = xr[tid + 256], v2 = xr[tid + 512];
  red[tid] = v0 + v1 + v2;
  __syncthreads();
  for (int s = 128; s > 0; s >>= 1) {
    if (tid < s) red[tid] += red[tid + s];
    __syncthreads();
  }
  float mean = red[0] * (1.0f / DD);
  __syncthreads();
  float d0 = v0 - mean, d1 = v1 - mean, d2 = v2 - mean;
  red[tid] = d0 * d0 + d1 * d1 + d2 * d2;
  __syncthreads();
  for (int s = 128; s > 0; s >>= 1) {
    if (tid < s) red[tid] += red[tid + s];
    __syncthreads();
  }
  float rstd = rsqrtf(red[0] * (1.0f / DD) + 1e-5f);
  unsigned short* o = out + (size_t)row * DD;
  o[tid]       = f2bf(d0 * rstd * w[tid]       + b[tid]);
  o[tid + 256] = f2bf(d1 * rstd * w[tid + 256] + b[tid + 256]);
  o[tid + 512] = f2bf(d2 * rstd * w[tid + 512] + b[tid + 512]);
}

// ---------------- final LN on cls rows -> d_out f32 -------------------------
__global__ __launch_bounds__(256) void ln_cls_out(const float* __restrict__ x,
                                                  const float* __restrict__ w,
                                                  const float* __restrict__ b,
                                                  float* __restrict__ out) {
  __shared__ float red[256];
  const int tid = threadIdx.x;
  const float* xr = x + (size_t)blockIdx.x * NN * DD;   // row 0 = cls
  float v0 = xr[tid], v1 = xr[tid + 256], v2 = xr[tid + 512];
  red[tid] = v0 + v1 + v2;
  __syncthreads();
  for (int s = 128; s > 0; s >>= 1) {
    if (tid < s) red[tid] += red[tid + s];
    __syncthreads();
  }
  float mean = red[0] * (1.0f / DD);
  __syncthreads();
  float d0 = v0 - mean, d1 = v1 - mean, d2 = v2 - mean;
  red[tid] = d0 * d0 + d1 * d1 + d2 * d2;
  __syncthreads();
  for (int s = 128; s > 0; s >>= 1) {
    if (tid < s) red[tid] += red[tid + s];
    __syncthreads();
  }
  float rstd = rsqrtf(red[0] * (1.0f / DD) + 1e-5f);
  float* o = out + (size_t)blockIdx.x * DD;
  o[tid]       = d0 * rstd * w[tid]       + b[tid];
  o[tid + 256] = d1 * rstd * w[tid + 256] + b[tid + 256];
  o[tid + 512] = d2 * rstd * w[tid + 512] + b[tid + 512];
}

// ---------------- tiled context pooling (2x2 tiles, U=1) --------------------
// grid = B*50; blockIdx%50 == 49 copies the cls row, else pools tile t.
__global__ __launch_bounds__(128) void ctx_pool(const unsigned short* __restrict__ xn,
                                                const unsigned short* __restrict__ lgw,
                                                unsigned short* __restrict__ ctx) {
  const int b = blockIdx.x / 50, t = blockIdx.x % 50, tid = threadIdx.x;
  const unsigned short* xb = xn + (size_t)b * NN * DD;
  unsigned short* cb = ctx + (size_t)b * KC * DD;
  if (t == 49) {                               // ctx[b,0] = xn[b,0] (cls)
    for (int c = tid; c < DD; c += 128) cb[c] = xb[c];
    return;
  }
  const int ti = t / 7, tj = t % 7;
  int tok[4];
#pragma unroll
  for (int s = 0; s < 4; ++s) {
    int a = s >> 1, w = s & 1;
    tok[s] = 1 + (2 * ti + a) * GRID + (2 * tj + w);
  }
  __shared__ float sred[4][128];
  float part[4] = {0.f, 0.f, 0.f, 0.f};
  for (int c = tid; c < DD; c += 128) {
    float lw = bf2f(lgw[c]);
#pragma unroll
    for (int s = 0; s < 4; ++s) part[s] += bf2f(xb[(size_t)tok[s] * DD + c]) * lw;
  }
#pragma unroll
  for (int s = 0; s < 4; ++s) sred[s][tid] = part[s];
  __syncthreads();
  for (int st = 64; st > 0; st >>= 1) {
    if (tid < st)
#pragma unroll
      for (int s = 0; s < 4; ++s) sred[s][tid] += sred[s][tid + st];
    __syncthreads();
  }
  __shared__ float wts[4];
  if (tid == 0) {
    float s0 = sred[0][0], s1 = sred[1][0], s2 = sred[2][0], s3 = sred[3][0];
    float mx = fmaxf(fmaxf(s0, s1), fmaxf(s2, s3));
    float e0 = __expf(s0 - mx), e1 = __expf(s1 - mx);
    float e2 = __expf(s2 - mx), e3 = __expf(s3 - mx);
    float inv = 1.0f / (e0 + e1 + e2 + e3);
    wts[0] = e0 * inv; wts[1] = e1 * inv; wts[2] = e2 * inv; wts[3] = e3 * inv;
  }
  __syncthreads();
  float w0 = wts[0], w1 = wts[1], w2 = wts[2], w3 = wts[3];
  for (int c = tid; c < DD; c += 128) {
    float v = w0 * bf2f(xb[(size_t)tok[0] * DD + c]) +
              w1 * bf2f(xb[(size_t)tok[1] * DD + c]) +
              w2 * bf2f(xb[(size_t)tok[2] * DD + c]) +
              w3 * bf2f(xb[(size_t)tok[3] * DD + c]);
    cb[(size_t)(1 + t) * DD + c] = f2bf(v);
  }
}

// ---------------- CPB relative-bias MLP: 2 -> 32 -> 12 ----------------------
__global__ void cpb_bias_k(const float* __restrict__ feats,
                           const float* __restrict__ mask,
                           const float* __restrict__ w1, const float* __restrict__ b1,
                           const float* __restrict__ w2, const float* __restrict__ b2,
                           float* __restrict__ bias) {
  int i = blockIdx.x * blockDim.x + threadIdx.x;
  if (i >= NN * KC) return;
  int n = i / KC, k = i % KC;
  float f0 = feats[(size_t)i * 2], f1 = feats[(size_t)i * 2 + 1];
  float h[32];
#pragma unroll
  for (int j = 0; j < 32; ++j)
    h[j] = gelu_exact(w1[j * 2] * f0 + w1[j * 2 + 1] * f1 + b1[j]);
  float m = mask[i];
  for (int hh = 0; hh < HH; ++hh) {
    float s = b2[hh];
#pragma unroll
    for (int j = 0; j < 32; ++j) s += w2[hh * 32 + j] * h[j];
    bias[((size_t)hh * NN + n) * KC + k] = s * m;
  }
}

// ---------------- attention: per (b,h); K/V resident in LDS -----------------
__global__ __launch_bounds__(256) void attn_k(const float* __restrict__ q,
                                              const float* __restrict__ kv,
                                              const float* __restrict__ bias,
                                              unsigned short* __restrict__ obuf) {
  const int b = blockIdx.x / HH, h = blockIdx.x % HH;
  __shared__ float ks[KC * DH];
  __shared__ float vs[KC * DH];
  const int tid = threadIdx.x;
  const float* kvb = kv + (size_t)b * KC * (2 * DD);
  for (int i = tid; i < KC * DH; i += 256) {
    int kk = i >> 6, dh = i & 63;
    ks[i] = kvb[(size_t)kk * (2 * DD) + h * DH + dh];
    vs[i] = kvb[(size_t)kk * (2 * DD) + DD + h * DH + dh];
  }
  __syncthreads();
  const int n = tid;
  if (n >= NN) return;
  const float* qr = q + ((size_t)(b * NN + n)) * DD + h * DH;
  float qv[DH];
#pragma unroll
  for (int d = 0; d < DH; ++d) qv[d] = qr[d];
  const float* bn = bias + ((size_t)h * NN + n) * KC;
  float lg[KC];
  float mx = -1e30f;
#pragma unroll
  for (int kk = 0; kk < KC; ++kk) {
    float s = 0.f;
#pragma unroll
    for (int d = 0; d < DH; ++d) s += qv[d] * ks[kk * DH + d];
    s = s * 0.125f + bn[kk];
    lg[kk] = s;
    mx = fmaxf(mx, s);
  }
  float sum = 0.f;
#pragma unroll
  for (int kk = 0; kk < KC; ++kk) {
    lg[kk] = __expf(lg[kk] - mx);
    sum += lg[kk];
  }
  float inv = 1.0f / sum;
  unsigned short* orow = obuf + ((size_t)(b * NN + n)) * DD + h * DH;
  for (int d = 0; d < DH; ++d) {
    float o = 0.f;
#pragma unroll
    for (int kk = 0; kk < KC; ++kk) o += lg[kk] * vs[kk * DH + d];
    orow[d] = f2bf(o * inv);
  }
}

// ---------------- host orchestration ----------------------------------------
static inline void cvt(const float* s, unsigned short* d, int n, hipStream_t st) {
  cvt_f32_bf16<<<(n + 255) / 256, 256, 0, st>>>(s, d, n);
}

template <int EPI>
static inline void gemm(const unsigned short* A, const unsigned short* W,
                        const float* bias, const float* res, const float* pos,
                        float* oF, unsigned short* oH, int M, int N, int K,
                        int rdiv, int roff, hipStream_t st) {
  dim3 g((M + 127) / 128, N / 64);
  gemm_bf16_wmma<EPI><<<g, 256, 0, st>>>(A, W, bias, res, pos, oF, oH, M, N, K, rdiv, roff);
}

extern "C" void kernel_launch(void* const* d_in, const int* in_sizes, int n_in,
                              void* d_out, int out_size, void* d_ws, size_t ws_size,
                              hipStream_t stream) {
  (void)in_sizes; (void)n_in; (void)out_size; (void)ws_size;
  const float* x_img   = (const float*)d_in[0];
  const float* patch_w = (const float*)d_in[1];
  const float* patch_b = (const float*)d_in[2];
  const float* cls_tok = (const float*)d_in[3];
  const float* pos_emb = (const float*)d_in[4];
  const float* n1_w    = (const float*)d_in[5];
  const float* n1_b    = (const float*)d_in[6];
  const float* logit_w = (const float*)d_in[7];
  const float* q_w     = (const float*)d_in[8];
  const float* kv_w    = (const float*)d_in[9];
  const float* out_w   = (const float*)d_in[10];
  const float* out_b   = (const float*)d_in[11];
  const float* cpb_w1  = (const float*)d_in[12];
  const float* cpb_b1  = (const float*)d_in[13];
  const float* cpb_w2  = (const float*)d_in[14];
  const float* cpb_b2  = (const float*)d_in[15];
  const float* n2_w    = (const float*)d_in[16];
  const float* n2_b    = (const float*)d_in[17];
  const float* fc1_w   = (const float*)d_in[18];
  const float* fc1_b   = (const float*)d_in[19];
  const float* fc2_w   = (const float*)d_in[20];
  const float* fc2_b   = (const float*)d_in[21];
  const float* fnorm_w = (const float*)d_in[22];
  const float* fnorm_b = (const float*)d_in[23];
  const float* feats   = (const float*)d_in[24];
  const float* maskp   = (const float*)d_in[25];

  char* ws = (char*)d_ws;
  size_t off = 0;
  auto take = [&](size_t bytes) -> char* {
    char* p = ws + off;
    off += (bytes + 255) & ~(size_t)255;
    return p;
  };
  float* xf    = (float*)take((size_t)BB * NN * DD * 4);          // residual stream
  float* qb    = (float*)take((size_t)BB * NN * DD * 4);          // Q
  float* kvb   = (float*)take((size_t)BB * KC * 2 * DD * 4);      // KV
  float* biasb = (float*)take((size_t)HH * NN * KC * 4);          // attn bias
  unsigned short* xn   = (unsigned short*)take((size_t)BB * NN * DD * 2);
  unsigned short* ctx  = (unsigned short*)take((size_t)BB * KC * DD * 2);
  unsigned short* obuf = (unsigned short*)take((size_t)BB * NN * DD * 2);
  unsigned short* h1   = (unsigned short*)take((size_t)BB * NN * HID * 2);
  unsigned short* pat  = (unsigned short*)take((size_t)BB * PP * DD * 2);
  unsigned short* wst  = (unsigned short*)take(W_TOT * 2);        // bf16 weight stage

  // ---- patch embed ----
  cvt(patch_w, wst, DD * DD, stream);
  {
    int tot = BB * PP * DD;
    im2col_patches<<<(tot + 255) / 256, 256, 0, stream>>>(x_img, pat, tot);
  }
  gemm<2>(pat, wst, patch_b, nullptr, pos_emb, xf, nullptr,
          BB * PP, DD, DD, PP, 1, stream);
  init_cls<<<(BB * DD + 255) / 256, 256, 0, stream>>>(cls_tok, pos_emb, xf);

  // ---- transformer blocks ----
  const int MROWS = BB * NN;   // 6304
  for (int l = 0; l < DEPTH; ++l) {
    cvt(q_w   + (size_t)l * DD * DD,        wst + W_Q,   DD * DD,        stream);
    cvt(kv_w  + (size_t)l * 2 * DD * DD,    wst + W_KV,  2 * DD * DD,    stream);
    cvt(out_w + (size_t)l * DD * DD,        wst + W_OUT, DD * DD,        stream);
    cvt(fc1_w + (size_t)l * HID * DD,       wst + W_FC1, HID * DD,       stream);
    cvt(fc2_w + (size_t)l * DD * HID,       wst + W_FC2, DD * HID,       stream);
    cvt(logit_w + (size_t)l * DD,           wst + W_LG,  DD,             stream);

    ln_bf16<<<MROWS, 256, 0, stream>>>(xf, n1_w + l * DD, n1_b + l * DD, xn);
    ctx_pool<<<BB * 50, 128, 0, stream>>>(xn, wst + W_LG, ctx);

    gemm<0>(xn,  wst + W_Q,  nullptr, nullptr, nullptr, qb,  nullptr,
            MROWS,   DD,     DD, 0, 0, stream);
    gemm<0>(ctx, wst + W_KV, nullptr, nullptr, nullptr, kvb, nullptr,
            BB * KC, 2 * DD, DD, 0, 0, stream);

    cpb_bias_k<<<(NN * KC + 255) / 256, 256, 0, stream>>>(
        feats, maskp, cpb_w1 + l * 64, cpb_b1 + l * 32,
        cpb_w2 + l * HH * 32, cpb_b2 + l * HH, biasb);

    attn_k<<<BB * HH, 256, 0, stream>>>(qb, kvb, biasb, obuf);

    gemm<1>(obuf, wst + W_OUT, out_b + l * DD, xf, nullptr, xf, nullptr,
            MROWS, DD, DD, 0, 0, stream);

    ln_bf16<<<MROWS, 256, 0, stream>>>(xf, n2_w + l * DD, n2_b + l * DD, xn);

    gemm<3>(xn, wst + W_FC1, fc1_b + (size_t)l * HID, nullptr, nullptr,
            nullptr, h1, MROWS, HID, DD, 0, 0, stream);
    gemm<1>(h1, wst + W_FC2, fc2_b + l * DD, xf, nullptr, xf, nullptr,
            MROWS, DD, HID, 0, 0, stream);
  }

  // ---- final LN on cls rows -> output ----
  ln_cls_out<<<BB, 256, 0, stream>>>(xf, fnorm_w, fnorm_b, (float*)d_out);
}
